// PGlobalMultiHeadAttention_23854248362061
// MI455X (gfx1250) — compile-verified, matrix operand order checked by reading, not run on hardware
//
#include <hip/hip_runtime.h>

// Problem constants
constexpr int kB   = 16;
constexpr int kL   = 512;
constexpr int kH   = 8;
constexpr int kD   = 128;   // d_k == d_v
constexpr int kDM  = 1024;  // d_model
constexpr int kM   = 3 * kB * kL;       // 24576 fused rows (mod,b,l)

typedef __attribute__((ext_vector_type(16))) _Float16 v16h;
typedef __attribute__((ext_vector_type(8)))  float    v8f;
typedef int b128v __attribute__((vector_size(4 * sizeof(int))));

#define WMMA_F16(A, B, C) \
  __builtin_amdgcn_wmma_f32_16x16x32_f16(false, (A), false, (B), (short)0, (C), false, false)

// ------------------------------------------------------------------
// CDNA5 async global->LDS copy (16 bytes per lane), ASYNCcnt tracked.
// Builtin if the toolchain declares it, otherwise inline gfx1250 asm.
// ------------------------------------------------------------------
#if defined(__has_builtin)
#if __has_builtin(__builtin_amdgcn_global_load_async_to_lds_b128)
#define USE_ASYNC_BUILTIN 1
#endif
#if __has_builtin(__builtin_amdgcn_s_wait_asynccnt)
#define USE_ASYNCWAIT_BUILTIN 1
#endif
#endif

__device__ __forceinline__ void async_cp16(_Float16* lds_dst, const _Float16* gsrc) {
#ifdef USE_ASYNC_BUILTIN
  // (global v4i* src, local v4i* dst, imm offset, imm cpol)
  __builtin_amdgcn_global_load_async_to_lds_b128(
      (__attribute__((address_space(1))) b128v*)(unsigned long long)(size_t)gsrc,
      (__attribute__((address_space(3))) b128v*)(unsigned)(size_t)lds_dst,
      0, 0);
#else
  // vdst = 32-bit LDS byte address, vaddr = 64-bit global address (GV mode)
  asm volatile("global_load_async_to_lds_b128 %0, %1, off"
               :: "v"((unsigned)(size_t)lds_dst),
                  "v"((unsigned long long)(size_t)gsrc)
               : "memory");
#endif
}

__device__ __forceinline__ void async_wait0() {
#ifdef USE_ASYNCWAIT_BUILTIN
  __builtin_amdgcn_s_wait_asynccnt(0);
#else
  asm volatile("s_wait_asynccnt 0x0" ::: "memory");
#endif
}

// Load a 16x32 f16 A-style fragment (also valid for B when the tile holds B^T
// row-major): lane (l&15) supplies matrix row, half (l>>4) selects K-chunks
// per the CDNA5 16-bit A layout (K = {half*8..+7, 16+half*8..+7}).
__device__ __forceinline__ v16h frag_ld(const _Float16* base, int stride, int lane) {
  const int r = lane & 15, hf = lane >> 4;
  const _Float16* p = base + r * stride + hf * 8;
  union { v16h v; uint4 q[2]; } f;
  f.q[0] = *(const uint4*)(p);
  f.q[1] = *(const uint4*)(p + 16);
  return f.v;
}

// ---------------------------------------------------------------- casts/prep
__global__ void cast_f32_f16(const float* __restrict__ x, _Float16* __restrict__ y, int n) {
  int i = blockIdx.x * blockDim.x + threadIdx.x;
  if (i < n) y[i] = (_Float16)x[i];
}

// w: [K=1024][N=1024] row-major f32 -> wt: [N][K] row-major f16
__global__ void transpose_cast_w(const float* __restrict__ w, _Float16* __restrict__ wt) {
  int i = blockIdx.x * blockDim.x + threadIdx.x;  // over 1M
  int nn = i >> 10, kk = i & 1023;
  wt[i] = (_Float16)w[kk * kDM + nn];
}

// ---------------------------------------------------------------- QKV GEMM
// X: [24576][1024] f16 row-major, Wt: [1024 n][1024 k] f16 (pre-transposed).
// VOUT==0: head-major f16 [3][B][H][L][128]   (Q and K)
// VOUT==1: d-major   f16 [3][B][H][128][L]    (V, pre-transposed for attention)
template <int VOUT>
__global__ __launch_bounds__(256)
void gemm_qkv(const _Float16* __restrict__ X, const _Float16* __restrict__ Wt,
              _Float16* __restrict__ out) {
  __shared__ _Float16 sA[64 * 40];    // 64 rows x BK=32 (+8 pad)
  __shared__ _Float16 sB[128 * 40];   // 128 B^T rows x 32 (+8 pad)
  const int t = threadIdx.x, lane = t & 31, wid = t >> 5;
  const int wm = wid & 3, wn = wid >> 2;          // waves: 4 along M, 2 along N
  const int rowBase = blockIdx.x * 64;
  const int colBase = blockIdx.y * 128;
  v8f acc[4] = {};

  const int ra = t >> 2, ca = (t & 3) * 8;        // A staging chunk
  for (int k0 = 0; k0 < kDM; k0 += 32) {
    async_cp16(&sA[ra * 40 + ca], &X[(size_t)(rowBase + ra) * kDM + k0 + ca]);
#pragma unroll
    for (int i = 0; i < 2; ++i) {  // stage B^T: 128 rows x 4 chunks = 512
      int cid = i * 256 + t;
      int r = cid >> 2, c = (cid & 3) * 8;
      async_cp16(&sB[r * 40 + c], &Wt[(size_t)(colBase + r) * kDM + k0 + c]);
    }
    if (k0 + 32 < kDM) {  // prefetch next tiles into GL2 while WMMAs run
      __builtin_prefetch(&X[(size_t)(rowBase + ra) * kDM + k0 + 32 + ca], 0, 1);
      __builtin_prefetch(&Wt[(size_t)(colBase + (t >> 1)) * kDM + k0 + 32], 0, 1);
    }
    async_wait0();
    __syncthreads();
    v16h a = frag_ld(&sA[wm * 16 * 40], 40, lane);
#pragma unroll
    for (int j = 0; j < 4; ++j) {
      v16h b = frag_ld(&sB[(wn * 64 + j * 16) * 40], 40, lane);
      acc[j] = WMMA_F16(a, b, acc[j]);
    }
    __syncthreads();
  }

  const int hf8 = (lane >> 4) * 8, cn = lane & 15;
#pragma unroll
  for (int j = 0; j < 4; ++j) {
    int gc = colBase + wn * 64 + j * 16 + cn;
    int h = gc >> 7, d = gc & 127;
#pragma unroll
    for (int r = 0; r < 8; ++r) {
      int gr  = rowBase + wm * 16 + hf8 + r;
      int mod = gr >> 13;            // / 8192
      int rem = gr & 8191;
      int b   = rem >> 9;            // / 512
      int l   = rem & 511;
      size_t head = ((size_t)(mod * kB + b) * kH + h);
      if (VOUT == 0)
        out[(head * kL + l) * kD + d] = (_Float16)acc[j][r];
      else
        out[(head * kD + d) * kL + l] = (_Float16)acc[j][r];
    }
  }
}

// ---------------------------------------------------------------- attention
// Qg/Kg head-major f16 [3][B][H][L][128]; Vt d-major f16 [3][B][H][128][L].
// Og: f16 [24576][1024] (row=(mod,b,l), col=h*128+d)
__global__ __launch_bounds__(128)
void attn_kernel(const _Float16* __restrict__ Qg, const _Float16* __restrict__ Kg,
                 const _Float16* __restrict__ Vt, _Float16* __restrict__ Og) {
  __shared__ _Float16 sQ[64 * 136];     // 64 q-rows x 128 (+8 pad)
  __shared__ _Float16 sK[32 * 136];     // 32 key rows x 128
  __shared__ _Float16 sVt[128 * 40];    // V^T: 128 d-rows x 32 keys (+8 pad)
  __shared__ _Float16 sP[4 * 16 * 40];  // per-wave P tile 16 q x 32 k

  const int t = threadIdx.x, lane = t & 31, wid = t >> 5;
  const int qt  = blockIdx.x & 7;
  const int h   = (blockIdx.x >> 3) & 7;
  const int b   = (blockIdx.x >> 6) & 15;
  const int mod = blockIdx.x >> 10;
  const int o1 = (mod == 0) ? 1 : 0;
  const int o2 = (mod == 2) ? 1 : 2;
  const size_t headQ  = ((size_t)((mod * kB + b) * kH + h)) * kL * kD;
  const size_t headK1 = ((size_t)((o1 * kB + b) * kH + h)) * kL * kD;
  const size_t headK2 = ((size_t)((o2 * kB + b) * kH + h)) * kL * kD;
  const int l0 = qt * 64;

  // stage Q tile (async): 64 rows x 16 chunks = 1024 chunks / 128 thr
#pragma unroll
  for (int i = 0; i < 8; ++i) {
    int cid = i * 128 + t;
    int r = cid >> 4, c = (cid & 15) * 8;
    async_cp16(&sQ[r * 136 + c], &Qg[headQ + (size_t)(l0 + r) * kD + c]);
  }
  async_wait0();
  __syncthreads();

  v16h qa[4];
#pragma unroll
  for (int dc = 0; dc < 4; ++dc)
    qa[dc] = frag_ld(&sQ[wid * 16 * 136 + dc * 32], 136, lane);

  v8f o[8] = {};
  float m_s[8], l_s[8];
#pragma unroll
  for (int r = 0; r < 8; ++r) { m_s[r] = -1e30f; l_s[r] = 0.f; }
  const float inv_t = 0.08838834764831845f;  // 1/sqrt(128)

  for (int kt = 0; kt < 32; ++kt) {
    const int gk0 = kt * 32;  // tiles never straddle the 512 boundary
    const int kk0 = (gk0 < 512) ? gk0 : gk0 - 512;
    const size_t kbase = ((gk0 < 512) ? headK1 : headK2) + (size_t)kk0 * kD;
    const size_t vbase = ((gk0 < 512) ? headK1 : headK2) + (size_t)kk0;  // d-major: + d*kL + key

#pragma unroll
    for (int i = 0; i < 4; ++i) {  // stage K: 32 rows x 16 chunks = 512
      int cid = i * 128 + t;
      int r = cid >> 4, c = (cid & 15) * 8;
      async_cp16(&sK[r * 136 + c], &Kg[kbase + (size_t)r * kD + c]);
    }
#pragma unroll
    for (int i = 0; i < 4; ++i) {  // stage V^T: 128 d-rows x 4 chunks(8 keys) = 512
      int cid = i * 128 + t;
      int dr = cid >> 2, c = (cid & 3) * 8;
      async_cp16(&sVt[dr * 40 + c], &Vt[vbase + (size_t)dr * kL + c]);
    }
    async_wait0();
    __syncthreads();

    // S tiles: 16q x 32k (two 16x16 WMMA groups, 4 d-chunks each)
    v8f s0 = {}, s1 = {};
#pragma unroll
    for (int dc = 0; dc < 4; ++dc) {
      v16h kb0 = frag_ld(&sK[0 * 136 + dc * 32], 136, lane);
      s0 = WMMA_F16(qa[dc], kb0, s0);
      v16h kb1 = frag_ld(&sK[16 * 136 + dc * 32], 136, lane);
      s1 = WMMA_F16(qa[dc], kb1, s1);
    }

    // online softmax (row m = 8*(lane>>4)+r, col = lane&15)
    _Float16* myP = &sP[wid * 16 * 40];
    const int hf8 = (lane >> 4) * 8, cn = lane & 15;
#pragma unroll
    for (int r = 0; r < 8; ++r) {
      float a0 = (float)s0[r] * inv_t, a1 = (float)s1[r] * inv_t;
      float mx = fmaxf(a0, a1);
#pragma unroll
      for (int msk = 1; msk < 16; msk <<= 1) mx = fmaxf(mx, __shfl_xor(mx, msk, 32));
      float mnew = fmaxf(m_s[r], mx);
      float corr = __expf(m_s[r] - mnew);
      float p0 = __expf(a0 - mnew), p1 = __expf(a1 - mnew);
      float rs = p0 + p1;
#pragma unroll
      for (int msk = 1; msk < 16; msk <<= 1) rs += __shfl_xor(rs, msk, 32);
      l_s[r] = l_s[r] * corr + rs;
      m_s[r] = mnew;
#pragma unroll
      for (int dt = 0; dt < 8; ++dt) o[dt][r] *= corr;
      myP[(hf8 + r) * 40 + cn]      = (_Float16)p0;
      myP[(hf8 + r) * 40 + 16 + cn] = (_Float16)p1;
    }
    __syncthreads();

    // O += P @ V : A = P (16x32), B^T rows = d (sVt)
    v16h pa = frag_ld(myP, 40, lane);
#pragma unroll
    for (int dt = 0; dt < 8; ++dt) {
      v16h vb = frag_ld(&sVt[dt * 16 * 40], 40, lane);
      o[dt] = WMMA_F16(pa, vb, o[dt]);
    }
    __syncthreads();
  }

  // normalize + store f16 [ (mod,b,l) ][ h*128 + d ]
  const int hf8 = (lane >> 4) * 8, cn = lane & 15;
#pragma unroll
  for (int r = 0; r < 8; ++r) {
    float inv_l = 1.f / l_s[r];
    int l = l0 + wid * 16 + hf8 + r;
    size_t outRow = ((size_t)(mod * kB + b) * kL + l) * kDM;
#pragma unroll
    for (int dt = 0; dt < 8; ++dt) {
      int col = h * kD + dt * 16 + cn;
      Og[outRow + col] = (_Float16)((float)o[dt][r] * inv_l);
    }
  }
}

// ---------------------------------------------------------------- FC + residual
__global__ __launch_bounds__(256)
void gemm_fc(const _Float16* __restrict__ A, const _Float16* __restrict__ Wt,
             const float* __restrict__ r0, const float* __restrict__ r1,
             const float* __restrict__ r2, float* __restrict__ out) {
  __shared__ _Float16 sA[64 * 40];
  __shared__ _Float16 sB[128 * 40];
  const int t = threadIdx.x, lane = t & 31, wid = t >> 5;
  const int wm = wid & 3, wn = wid >> 2;
  const int rowBase = blockIdx.x * 64;
  const int colBase = blockIdx.y * 128;
  v8f acc[4] = {};

  const int ra = t >> 2, ca = (t & 3) * 8;
  for (int k0 = 0; k0 < kDM; k0 += 32) {
    async_cp16(&sA[ra * 40 + ca], &A[(size_t)(rowBase + ra) * kDM + k0 + ca]);
#pragma unroll
    for (int i = 0; i < 2; ++i) {
      int cid = i * 256 + t;
      int r = cid >> 2, c = (cid & 3) * 8;
      async_cp16(&sB[r * 40 + c], &Wt[(size_t)(colBase + r) * kDM + k0 + c]);
    }
    if (k0 + 32 < kDM) {
      __builtin_prefetch(&A[(size_t)(rowBase + ra) * kDM + k0 + 32 + ca], 0, 1);
      __builtin_prefetch(&Wt[(size_t)(colBase + (t >> 1)) * kDM + k0 + 32], 0, 1);
    }
    async_wait0();
    __syncthreads();
    v16h a = frag_ld(&sA[wm * 16 * 40], 40, lane);
#pragma unroll
    for (int j = 0; j < 4; ++j) {
      v16h b = frag_ld(&sB[(wn * 64 + j * 16) * 40], 40, lane);
      acc[j] = WMMA_F16(a, b, acc[j]);
    }
    __syncthreads();
  }

  const int hf8 = (lane >> 4) * 8, cn = lane & 15;
#pragma unroll
  for (int j = 0; j < 4; ++j) {
    int gc = colBase + wn * 64 + j * 16 + cn;
#pragma unroll
    for (int r = 0; r < 8; ++r) {
      int gr  = rowBase + wm * 16 + hf8 + r;
      int mod = gr >> 13;
      int rem = gr & 8191;
      const float* res = (mod == 0) ? r0 : (mod == 1) ? r1 : r2;
      out[(size_t)gr * kDM + gc] = acc[j][r] + res[(size_t)rem * kDM + gc];
    }
  }
}

// ---------------------------------------------------------------- layernorm (in place)
__global__ __launch_bounds__(256)
void layernorm_inplace(float* __restrict__ y, const float* __restrict__ g,
                       const float* __restrict__ bia) {
  const int row = blockIdx.x, t = threadIdx.x;
  const int lane = t & 31, wid = t >> 5;
  float v[4], s = 0.f, s2 = 0.f;
#pragma unroll
  for (int i = 0; i < 4; ++i) {
    v[i] = y[(size_t)row * kDM + t + i * 256];
    s += v[i]; s2 += v[i] * v[i];
  }
#pragma unroll
  for (int m = 1; m < 32; m <<= 1) { s += __shfl_xor(s, m, 32); s2 += __shfl_xor(s2, m, 32); }
  __shared__ float rs[8], rs2[8];
  __shared__ float mean_sh, rstd_sh;
  if (lane == 0) { rs[wid] = s; rs2[wid] = s2; }
  __syncthreads();
  if (t == 0) {
    float S = 0.f, S2 = 0.f;
#pragma unroll
    for (int i = 0; i < 8; ++i) { S += rs[i]; S2 += rs2[i]; }
    float mu = S / kDM, var = S2 / kDM - mu * mu;
    mean_sh = mu; rstd_sh = rsqrtf(var + 1e-6f);
  }
  __syncthreads();
  float mu = mean_sh, rstd = rstd_sh;
#pragma unroll
  for (int i = 0; i < 4; ++i) {
    int c = t + i * 256;
    y[(size_t)row * kDM + c] = (v[i] - mu) * rstd * g[c] + bia[c];
  }
}

// ---------------------------------------------------------------- launcher
extern "C" void kernel_launch(void* const* d_in, const int* in_sizes, int n_in,
                              void* d_out, int out_size, void* d_ws, size_t ws_size,
                              hipStream_t stream) {
  (void)in_sizes; (void)n_in; (void)out_size; (void)ws_size;
  const float* in_t = (const float*)d_in[0];
  const float* in_a = (const float*)d_in[1];
  const float* in_v = (const float*)d_in[2];
  const float* w_qs = (const float*)d_in[3];
  const float* w_ks = (const float*)d_in[4];
  const float* w_vs = (const float*)d_in[5];
  const float* w_fc = (const float*)d_in[6];
  const float* ln_g = (const float*)d_in[7];
  const float* ln_b = (const float*)d_in[8];
  float* outp = (float*)d_out;

  // workspace layout (f16 elements)
  const size_t nX = (size_t)kM * kDM;       // 25,165,824 per tensor
  const size_t nW = (size_t)kDM * kDM;      // 1,048,576
  char* ws = (char*)d_ws;
  _Float16* Xf  = (_Float16*)(ws);
  _Float16* WqT = Xf + nX;
  _Float16* WkT = WqT + nW;
  _Float16* WvT = WkT + nW;
  _Float16* WfT = WvT + nW;
  _Float16* Qf  = WfT + nW;
  _Float16* Kf  = Qf + nX;
  _Float16* Vf  = Kf + nX;   // d-major (pre-transposed)
  _Float16* AOf = Vf + nX;

  const int nPer = kB * kL * kDM;  // 8,388,608 per input
  cast_f32_f16<<<nPer / 256, 256, 0, stream>>>(in_t, Xf + 0 * (size_t)nPer, nPer);
  cast_f32_f16<<<nPer / 256, 256, 0, stream>>>(in_a, Xf + 1 * (size_t)nPer, nPer);
  cast_f32_f16<<<nPer / 256, 256, 0, stream>>>(in_v, Xf + 2 * (size_t)nPer, nPer);

  transpose_cast_w<<<(int)(nW / 256), 256, 0, stream>>>(w_qs, WqT);
  transpose_cast_w<<<(int)(nW / 256), 256, 0, stream>>>(w_ks, WkT);
  transpose_cast_w<<<(int)(nW / 256), 256, 0, stream>>>(w_vs, WvT);
  transpose_cast_w<<<(int)(nW / 256), 256, 0, stream>>>(w_fc, WfT);

  dim3 gGemm(kM / 64, kDM / 128);
  gemm_qkv<0><<<gGemm, 256, 0, stream>>>(Xf, WqT, Qf);
  gemm_qkv<0><<<gGemm, 256, 0, stream>>>(Xf, WkT, Kf);
  gemm_qkv<1><<<gGemm, 256, 0, stream>>>(Xf, WvT, Vf);

  attn_kernel<<<3 * kB * kH * (kL / 64), 128, 0, stream>>>(Qf, Kf, Vf, AOf);

  gemm_fc<<<gGemm, 256, 0, stream>>>(AOf, WfT, in_t, in_a, in_v, outp);

  layernorm_inplace<<<kM, 256, 0, stream>>>(outp, ln_g, ln_b);
}
